// Swin2SRLayer_28381143892672
// MI455X (gfx1250) — matrix operand order repaired and checked by reading, not compiled, not run on hardware
//
#include <hip/hip_runtime.h>
#include <hip/hip_bf16.h>

typedef __attribute__((ext_vector_type(16))) _Float16 v16h;
typedef __attribute__((ext_vector_type(8)))  _Float16 v8h;
typedef __attribute__((ext_vector_type(8)))  float    v8f;

#define SX  200   // f16 LDS row stride (halves), multiple of 8 -> 16B aligned frags
#define SCF 66    // scores (f32) row stride, scalar access only
#define SPH 72    // probs (f16) row stride, multiple of 8
#define MS  392   // f16 LDS row stride for 384-wide MLP hidden (multiple of 8)
#define FS  196   // f32 LDS row stride for mlp output (scalar access)

// ---------------- WMMA fragment loaders (CDNA5 16x16x32 f16 layouts) --------
// A (16xK slice, MxK): lane m=lane&15 row; hi=lane>>4; halves K=i+16*hi
// (16 contiguous halves at k0+16*hi)
__device__ __forceinline__ v16h load_frag_A(const _Float16* base, int stride, int lane) {
    int m = lane & 15, hi = lane >> 4;
    const _Float16* p = base + m * stride + 16 * hi;
    v8h lo = *(const v8h*)(p);
    v8h hh = *(const v8h*)(p + 8);
    return __builtin_shufflevector(lo, hh, 0, 1, 2, 3, 4, 5, 6, 7,
                                           8, 9, 10, 11, 12, 13, 14, 15);
}
// B (KxN) from an [N][K]-stored matrix: lane n=lane&15 col; halves K=i+16*hi
// -> 16 contiguous halves (two 128-bit loads).
__device__ __forceinline__ v16h load_frag_B_T(const _Float16* wt, int strideK,
                                              int n0, int k0, int lane) {
    int n = lane & 15, hi = lane >> 4;
    const _Float16* p = wt + (long)(n0 + n) * strideK + k0 + 16 * hi;
    v8h lo = *(const v8h*)(p);
    v8h hh = *(const v8h*)(p + 8);
    return __builtin_shufflevector(lo, hh, 0, 1, 2, 3, 4, 5, 6, 7,
                                           8, 9, 10, 11, 12, 13, 14, 15);
}
// B (KxN) from a row-major [K][N] matrix (strided, scalar loads).
__device__ __forceinline__ v16h load_frag_B_rm(const _Float16* base, int stride,
                                               int k0, int n0, int lane) {
    int n = lane & 15, hi = lane >> 4;
    const _Float16* p = base + (k0 + 16 * hi) * stride + n0 + n;
    v16h b;
#pragma unroll
    for (int i = 0; i < 16; ++i) b[i] = p[i * stride];
    return b;
}

// ---------------- weight packing: f32 row-major -> f16 [N][K] ---------------
__global__ void pack_weights(const float* __restrict__ qw, const float* __restrict__ kw,
                             const float* __restrict__ vw, const float* __restrict__ pw,
                             const float* __restrict__ f1, const float* __restrict__ f2,
                             _Float16* __restrict__ wqkv, _Float16* __restrict__ pjt,
                             _Float16* __restrict__ f1t, _Float16* __restrict__ f2t) {
    const int total = 576 * 192 + 192 * 192 + 384 * 192 + 192 * 384;
    for (int e = blockIdx.x * blockDim.x + threadIdx.x; e < total; e += gridDim.x * blockDim.x) {
        int i = e;
        if (i < 576 * 192) {
            int n = i / 192, k = i % 192;
            float v = (n < 192) ? qw[k * 192 + n]
                    : (n < 384) ? kw[k * 192 + (n - 192)]
                                : vw[k * 192 + (n - 384)];
            wqkv[n * 192 + k] = (_Float16)v;
        } else {
            i -= 576 * 192;
            if (i < 192 * 192) {
                int n = i / 192, k = i % 192;
                pjt[n * 192 + k] = (_Float16)pw[k * 192 + n];
            } else {
                i -= 192 * 192;
                if (i < 384 * 192) {
                    int n = i / 192, k = i % 192;
                    f1t[n * 192 + k] = (_Float16)f1[k * 384 + n];
                } else {
                    i -= 384 * 192;
                    int n = i / 384, k = i % 384;
                    f2t[n * 384 + k] = (_Float16)f2[k * 192 + n];
                }
            }
        }
    }
}

// ---------------- continuous relative position bias ------------------------
__global__ void rpb_kernel(const float* __restrict__ w1, const float* __restrict__ b1,
                           const float* __restrict__ w2, float* __restrict__ rpb) {
    __shared__ float tab[225][6];
    int tid = threadIdx.x;
    if (tid < 225) {
        int i = tid / 15, j = tid % 15;
        float y0 = (float)(i - 7) * (8.0f / 7.0f);
        float y1 = (float)(j - 7) * (8.0f / 7.0f);
        float t0 = ((y0 > 0.f) ? 1.f : (y0 < 0.f) ? -1.f : 0.f) * log2f(fabsf(y0) + 1.0f) * (1.0f / 3.0f);
        float t1 = ((y1 > 0.f) ? 1.f : (y1 < 0.f) ? -1.f : 0.f) * log2f(fabsf(y1) + 1.0f) * (1.0f / 3.0f);
        float acc[6] = {0.f, 0.f, 0.f, 0.f, 0.f, 0.f};
        for (int u = 0; u < 512; ++u) {
            float hv = fmaxf(t0 * w1[u] + t1 * w1[512 + u] + b1[u], 0.0f);
#pragma unroll
            for (int h = 0; h < 6; ++h) acc[h] += hv * w2[u * 6 + h];
        }
#pragma unroll
        for (int h = 0; h < 6; ++h) tab[tid][h] = acc[h];
    }
    __syncthreads();
    for (int e = tid; e < 6 * 64 * 64; e += blockDim.x) {
        int h = e / 4096, a = (e / 64) % 64, b = e % 64;
        int d0 = (a >> 3) - (b >> 3) + 7;
        int d1 = (a & 7) - (b & 7) + 7;
        float v = tab[d0 * 15 + d1][h];
        rpb[e] = 16.0f / (1.0f + expf(-v));
    }
}

// ---------------- fused shifted-window attention (one block per window) -----
__global__ void __launch_bounds__(256) attn_kernel(
    const float* __restrict__ hidden, const float* __restrict__ qb,
    const float* __restrict__ vb, const float* __restrict__ ls,
    const _Float16* __restrict__ wqkv, const _Float16* __restrict__ pjt,
    const float* __restrict__ pb, const float* __restrict__ rpb,
    float* __restrict__ out) {
    __shared__ __align__(16) _Float16 Xs[64 * SX];
    __shared__ __align__(16) _Float16 qs[64 * SX];
    __shared__ __align__(16) _Float16 ks[64 * SX];
    __shared__ __align__(16) _Float16 vs[64 * SX];
    __shared__ __align__(16) _Float16 cs[64 * SX];
    __shared__ __align__(16) float    scr[64 * SCF];
    __shared__ __align__(16) _Float16 prb[64 * SPH];
    __shared__ __align__(16) float    sqc[64][12];
    __shared__ __align__(16) float    skc[64][12];
    __shared__ __align__(16) float    rqa[64];
    __shared__ __align__(16) float    rka[64];
    __shared__ __align__(16) int      rowptr[64];
    __shared__ __align__(16) int      regid[64];
    __shared__ __align__(16) float    scl[6];

    int tid = threadIdx.x, lane = tid & 31, wid = tid >> 5;
    int w = blockIdx.x, wy = w / 48, wx = w % 48;

    if (tid < 64) {
        int i = tid >> 3, j = tid & 7;
        int oy = wy * 8 + i + 4; if (oy >= 384) oy -= 384;
        int ox = wx * 8 + j + 4; if (ox >= 384) ox -= 384;
        rowptr[tid] = oy * 384 + ox;
        int ry = (wy == 47) ? ((i < 4) ? 1 : 2) : 0;
        int rx = (wx == 47) ? ((j < 4) ? 1 : 2) : 0;
        regid[tid] = ry * 3 + rx;
    }
    if (tid < 6) scl[tid] = expf(fminf(ls[tid], 4.6051702f));
    __syncthreads();

    for (int e = tid; e < 64 * 192; e += 256) {
        int t = e / 192, c = e % 192;
        Xs[t * SX + c] = (_Float16)hidden[(long)rowptr[t] * 192 + c];
    }
    __syncthreads();

    // ---- QKV: [64x192] @ [192x576], 144 wave-tiles (mt fixed per wave) ----
    for (int t = wid; t < 144; t += 8) {
        int mt = t & 3, nt = t >> 2;
        v8f acc = {};
#pragma unroll
        for (int k = 0; k < 6; ++k) {
            v16h a = load_frag_A(Xs + mt * 16 * SX + k * 32, SX, lane);
            v16h b = load_frag_B_T(wqkv, 192, nt * 16, k * 32, lane);
            acc = __builtin_amdgcn_wmma_f32_16x16x32_f16(false, a, false, b, (short)0, acc, false, false);
        }
        int nl = lane & 15, hi = lane >> 4;
        if (nt < 12) {                      // Q (+bias, sumsq)
            float bias = qb[nt * 16 + nl];
#pragma unroll
            for (int g = 0; g < 8; ++g) {
                int row = mt * 16 + g + 8 * hi;
                float v = acc[g] + bias;
                float ss = v * v;
#pragma unroll
                for (int o = 8; o > 0; o >>= 1) ss += __shfl_xor(ss, o, 16);
                if (nl == 0) sqc[row][nt] = ss;
                qs[row * SX + nt * 16 + nl] = (_Float16)v;
            }
        } else if (nt < 24) {               // K (sumsq)
            int ntk = nt - 12;
#pragma unroll
            for (int g = 0; g < 8; ++g) {
                int row = mt * 16 + g + 8 * hi;
                float v = acc[g];
                float ss = v * v;
#pragma unroll
                for (int o = 8; o > 0; o >>= 1) ss += __shfl_xor(ss, o, 16);
                if (nl == 0) skc[row][ntk] = ss;
                ks[row * SX + ntk * 16 + nl] = (_Float16)v;
            }
        } else {                            // V (+bias)
            int ntv = nt - 24;
            float bias = vb[ntv * 16 + nl];
#pragma unroll
            for (int g = 0; g < 8; ++g) {
                int row = mt * 16 + g + 8 * hi;
                vs[row * SX + ntv * 16 + nl] = (_Float16)(acc[g] + bias);
            }
        }
    }
    __syncthreads();

    // ---- per-head: scores -> softmax -> ctx ----
    for (int h = 0; h < 6; ++h) {
        if (tid < 64) {
            rqa[tid] = 1.0f / fmaxf(sqrtf(sqc[tid][2 * h] + sqc[tid][2 * h + 1]), 1e-12f);
        } else if (tid < 128) {
            int t = tid - 64;
            rka[t] = 1.0f / fmaxf(sqrtf(skc[t][2 * h] + skc[t][2 * h + 1]), 1e-12f);
        }
        __syncthreads();
        float sc = scl[h];
        // scores: 16 tiles, 2 per wave, K=32 (one WMMA)
        for (int t = wid * 2; t < wid * 2 + 2; ++t) {
            int mt = t >> 2, nt = t & 3;
            v16h a = load_frag_A(qs + mt * 16 * SX + h * 32, SX, lane);
            v16h b = load_frag_B_T(ks, SX, nt * 16, h * 32, lane);
            v8f acc = {};
            acc = __builtin_amdgcn_wmma_f32_16x16x32_f16(false, a, false, b, (short)0, acc, false, false);
            int nl = lane & 15, hi = lane >> 4;
            int bcol = nt * 16 + nl;
            float rkb = rka[bcol];
            int rgb = regid[bcol];
#pragma unroll
            for (int g = 0; g < 8; ++g) {
                int arow = mt * 16 + g + 8 * hi;
                float s = acc[g] * sc * rqa[arow] * rkb;
                s += rpb[(h * 64 + arow) * 64 + bcol];
                if (regid[arow] != rgb) s -= 200.0f;   // 2.0 * (-100) mask
                scr[arow * SCF + bcol] = s;
            }
        }
        __syncthreads();
        // softmax over 64 keys (one thread per row)
        if (tid < 64) {
            const float* r = scr + tid * SCF;
            float mx = -1e30f;
            for (int m = 0; m < 64; ++m) mx = fmaxf(mx, r[m]);
            float sum = 0.0f;
            for (int m = 0; m < 64; ++m) sum += __expf(r[m] - mx);
            float inv = 1.0f / sum;
            _Float16* pr = prb + tid * SPH;
            for (int m = 0; m < 64; ++m) pr[m] = (_Float16)(__expf(r[m] - mx) * inv);
        }
        __syncthreads();
        // ctx: 8 tiles, 1 per wave, K=64 (2 WMMAs)
        {
            int mt = wid >> 1, nt = wid & 1;
            v8f acc = {};
#pragma unroll
            for (int kk = 0; kk < 2; ++kk) {
                v16h a = load_frag_A(prb + mt * 16 * SPH + kk * 32, SPH, lane);
                v16h b = load_frag_B_rm(vs, SX, kk * 32, h * 32 + nt * 16, lane);
                acc = __builtin_amdgcn_wmma_f32_16x16x32_f16(false, a, false, b, (short)0, acc, false, false);
            }
            int nl = lane & 15, hi = lane >> 4;
#pragma unroll
            for (int g = 0; g < 8; ++g)
                cs[(mt * 16 + g + 8 * hi) * SX + h * 32 + nt * 16 + nl] = (_Float16)acc[g];
        }
        __syncthreads();
    }

    // ---- proj: [64x192] @ [192x192] + bias, scatter to unshifted rows ----
    for (int t = wid; t < 48; t += 8) {
        int mt = t & 3, nt = t >> 2;
        v8f acc = {};
#pragma unroll
        for (int k = 0; k < 6; ++k) {
            v16h a = load_frag_A(cs + mt * 16 * SX + k * 32, SX, lane);
            v16h b = load_frag_B_T(pjt, 192, nt * 16, k * 32, lane);
            acc = __builtin_amdgcn_wmma_f32_16x16x32_f16(false, a, false, b, (short)0, acc, false, false);
        }
        int nl = lane & 15, hi = lane >> 4;
        int col = nt * 16 + nl;
        float bias = pb[col];
#pragma unroll
        for (int g = 0; g < 8; ++g) {
            int trow = mt * 16 + g + 8 * hi;
            out[(long)rowptr[trow] * 192 + col] = acc[g] + bias;
        }
    }
}

// ---------------- h = shortcut + LN(x), in-place on x ----------------------
__global__ void __launch_bounds__(256) ln1_kernel(float* __restrict__ x,
                                                  const float* __restrict__ shortcut,
                                                  const float* __restrict__ g,
                                                  const float* __restrict__ b) {
    int lane = threadIdx.x & 31, wid = threadIdx.x >> 5;
    long row = (long)blockIdx.x * 8 + wid;
    float* xr = x + row * 192;
    float v[6];
    float s = 0.f, s2 = 0.f;
#pragma unroll
    for (int j = 0; j < 6; ++j) {
        v[j] = xr[lane + 32 * j];
        s += v[j]; s2 += v[j] * v[j];
    }
#pragma unroll
    for (int o = 16; o > 0; o >>= 1) { s += __shfl_xor(s, o, 32); s2 += __shfl_xor(s2, o, 32); }
    float mu = s * (1.0f / 192.0f);
    float var = s2 * (1.0f / 192.0f) - mu * mu;
    float rs = rsqrtf(var + 1e-5f);
    const float* sr = shortcut + row * 192;
#pragma unroll
    for (int j = 0; j < 6; ++j) {
        int c = lane + 32 * j;
        xr[c] = sr[c] + (v[j] - mu) * rs * g[c] + b[c];
    }
}

// ---------------- fused MLP + LN2 + residual, in-place ----------------------
__global__ void __launch_bounds__(256) mlp_kernel(float* __restrict__ io,
                                                  const _Float16* __restrict__ f1t,
                                                  const float* __restrict__ f1b,
                                                  const _Float16* __restrict__ f2t,
                                                  const float* __restrict__ f2b,
                                                  const float* __restrict__ g2,
                                                  const float* __restrict__ b2) {
    __shared__ __align__(16) _Float16 hs[64 * SX];
    __shared__ __align__(16) _Float16 mid[64 * MS];
    __shared__ __align__(16) float    mlpf[64 * FS];
    int tid = threadIdx.x, lane = tid & 31, wid = tid >> 5;
    long base = (long)blockIdx.x * 64;

    for (int e = tid; e < 64 * 192; e += 256) {
        int t = e / 192, c = e % 192;
        hs[t * SX + c] = (_Float16)io[(base + t) * 192 + c];
    }
    __syncthreads();

    // fc1 + GELU: [64x192] @ [192x384]
    for (int t = wid; t < 96; t += 8) {
        int mt = t & 3, nt = t >> 2;
        v8f acc = {};
#pragma unroll
        for (int k = 0; k < 6; ++k) {
            v16h a = load_frag_A(hs + mt * 16 * SX + k * 32, SX, lane);
            v16h b = load_frag_B_T(f1t, 192, nt * 16, k * 32, lane);
            acc = __builtin_amdgcn_wmma_f32_16x16x32_f16(false, a, false, b, (short)0, acc, false, false);
        }
        int nl = lane & 15, hi = lane >> 4;
        float bias = f1b[nt * 16 + nl];
#pragma unroll
        for (int g = 0; g < 8; ++g) {
            float xg = acc[g] + bias;
            float ge = 0.5f * xg * (1.0f + erff(xg * 0.70710678f));
            mid[(mt * 16 + g + 8 * hi) * MS + nt * 16 + nl] = (_Float16)ge;
        }
    }
    __syncthreads();

    // fc2: [64x384] @ [384x192]
    for (int t = wid; t < 48; t += 8) {
        int mt = t & 3, nt = t >> 2;
        v8f acc = {};
#pragma unroll
        for (int k = 0; k < 12; ++k) {
            v16h a = load_frag_A(mid + mt * 16 * MS + k * 32, MS, lane);
            v16h b = load_frag_B_T(f2t, 384, nt * 16, k * 32, lane);
            acc = __builtin_amdgcn_wmma_f32_16x16x32_f16(false, a, false, b, (short)0, acc, false, false);
        }
        int nl = lane & 15, hi = lane >> 4;
        float bias = f2b[nt * 16 + nl];
#pragma unroll
        for (int g = 0; g < 8; ++g)
            mlpf[(mt * 16 + g + 8 * hi) * FS + nt * 16 + nl] = acc[g] + bias;
    }
    __syncthreads();

    // out = h + LN(mlp), in-place
    for (int r = wid; r < 64; r += 8) {
        const float* mr = mlpf + r * FS;
        float v[6];
        float s = 0.f, s2 = 0.f;
#pragma unroll
        for (int j = 0; j < 6; ++j) {
            v[j] = mr[lane + 32 * j];
            s += v[j]; s2 += v[j] * v[j];
        }
#pragma unroll
        for (int o = 16; o > 0; o >>= 1) { s += __shfl_xor(s, o, 32); s2 += __shfl_xor(s2, o, 32); }
        float mu = s * (1.0f / 192.0f);
        float var = s2 * (1.0f / 192.0f) - mu * mu;
        float rs = rsqrtf(var + 1e-5f);
        float* orow = io + (base + r) * 192;
#pragma unroll
        for (int j = 0; j < 6; ++j) {
            int c = lane + 32 * j;
            orow[c] = orow[c] + (v[j] - mu) * rs * g2[c] + b2[c];
        }
    }
}

extern "C" void kernel_launch(void* const* d_in, const int* in_sizes, int n_in,
                              void* d_out, int out_size, void* d_ws, size_t ws_size,
                              hipStream_t stream) {
    const float* hidden = (const float*)d_in[0];
    const float* q_w    = (const float*)d_in[1];
    const float* q_b    = (const float*)d_in[2];
    const float* k_w    = (const float*)d_in[3];
    const float* v_w    = (const float*)d_in[4];
    const float* v_b    = (const float*)d_in[5];
    const float* lscale = (const float*)d_in[6];
    const float* cpb_w1 = (const float*)d_in[7];
    const float* cpb_b1 = (const float*)d_in[8];
    const float* cpb_w2 = (const float*)d_in[9];
    const float* proj_w = (const float*)d_in[10];
    const float* proj_b = (const float*)d_in[11];
    const float* ln1_g  = (const float*)d_in[12];
    const float* ln1_b  = (const float*)d_in[13];
    const float* fc1_w  = (const float*)d_in[14];
    const float* fc1_b  = (const float*)d_in[15];
    const float* fc2_w  = (const float*)d_in[16];
    const float* fc2_b  = (const float*)d_in[17];
    const float* ln2_g  = (const float*)d_in[18];
    const float* ln2_b  = (const float*)d_in[19];
    float* out = (float*)d_out;

    char* ws = (char*)d_ws;
    _Float16* wqkv = (_Float16*)ws; ws += (size_t)576 * 192 * 2;
    _Float16* pjt  = (_Float16*)ws; ws += (size_t)192 * 192 * 2;
    _Float16* f1t  = (_Float16*)ws; ws += (size_t)384 * 192 * 2;
    _Float16* f2t  = (_Float16*)ws; ws += (size_t)192 * 384 * 2;
    float*    rpb  = (float*)ws;

    pack_weights<<<128, 256, 0, stream>>>(q_w, k_w, v_w, proj_w, fc1_w, fc2_w,
                                          wqkv, pjt, f1t, f2t);
    rpb_kernel<<<1, 256, 0, stream>>>(cpb_w1, cpb_b1, cpb_w2, rpb);
    attn_kernel<<<2304, 256, 0, stream>>>(hidden, q_b, v_b, lscale,
                                          wqkv, pjt, proj_b, rpb, out);
    ln1_kernel<<<18432, 256, 0, stream>>>(out, hidden, ln1_g, ln1_b);
    mlp_kernel<<<2304, 256, 0, stream>>>(out, f1t, fc1_b, f2t, fc2_b, ln2_g, ln2_b);
}
